// PhaseCoupledComplexAttention_24902220382567
// MI455X (gfx1250) — matrix operand order repaired and checked by reading, not compiled, not run on hardware
//
#include <hip/hip_runtime.h>
#include <hip/hip_bf16.h>

// ---------------------------------------------------------------------------
// Types
// ---------------------------------------------------------------------------
typedef __attribute__((ext_vector_type(16))) __bf16 v16bf;
typedef __attribute__((ext_vector_type(8)))  float  v8f;
typedef int v4i __attribute__((vector_size(16)));

struct U8 { unsigned int x[8]; };
static __device__ inline v16bf as_bf16x16(const U8& u) {
    return __builtin_bit_cast(v16bf, u);
}

static __device__ inline v8f wmma_bf16(v16bf a, v16bf b, v8f c) {
    // D = A(16x32 bf16) * B(32x16 bf16) + C(16x16 f32)
    return __builtin_amdgcn_wmma_f32_16x16x32_bf16(
        /*neg_a=*/false, a, /*neg_b=*/false, b,
        /*c_mod=*/(short)0, c, /*reuse_a=*/false, /*reuse_b=*/false);
}

// round-to-nearest-even f32 -> bf16
static __device__ inline unsigned short f2bf(float f) {
    unsigned int u = __builtin_bit_cast(unsigned int, f);
    u += 0x7FFFu + ((u >> 16) & 1u);
    return (unsigned short)(u >> 16);
}

// ---------------------------------------------------------------------------
// Optional async global->LDS DMA path (GLOBAL_LOAD_ASYNC_TO_LDS_B128, ASYNCcnt)
// ---------------------------------------------------------------------------
#if defined(__has_builtin)
#if __has_builtin(__builtin_amdgcn_global_load_async_to_lds_b128) && \
    __has_builtin(__builtin_amdgcn_s_wait_asynccnt)
#define USE_ASYNC_LDS 1
#endif
#endif
#ifndef USE_ASYNC_LDS
#define USE_ASYNC_LDS 0
#endif

#if USE_ASYNC_LDS
static __device__ inline void async_copy_b128(const void* gsrc, void* ldst) {
    __builtin_amdgcn_global_load_async_to_lds_b128(
        (__attribute__((address_space(1))) v4i*)(gsrc),
        (__attribute__((address_space(3))) v4i*)(ldst),
        /*imm offset=*/0, /*cpol=*/0);
}
static __device__ inline void async_wait0() {
    __builtin_amdgcn_s_wait_asynccnt(0);
}
#endif

// ---------------------------------------------------------------------------
// Problem dims
// ---------------------------------------------------------------------------
#define DIM   1024
#define NHEAD 16
#define HDIM  64
#define BATCH 2
#define SEQ   2048
#define ROWS  (BATCH * SEQ)   // 4096
#define K2    (2 * DIM)       // 2048

// ---------------------------------------------------------------------------
// Pack kernels
// ---------------------------------------------------------------------------
// Zc[m, k] = (k < D) ? zr[m,k] : zi[m,k-D], bf16 row-major [ROWS x 2D]
__global__ __launch_bounds__(256) void pack_z_kernel(const float* __restrict__ zr,
                                                     const float* __restrict__ zi,
                                                     unsigned short* __restrict__ zc) {
    int idx = blockIdx.x * 256 + threadIdx.x;          // over ROWS*K2
    int m = idx >> 11;
    int k = idx & (K2 - 1);
    float v = (k < DIM) ? zr[m * DIM + k] : zi[m * DIM + (k - DIM)];
    zc[idx] = f2bf(v);
}

// Combined, transposed, K-pair-interleaved weights:
//   WcR[k,n] = (k<D) ?  wr[n,k] : -wi[n,k-D]
//   WcI[k,n] = (k<D) ?  wi[n,k] :  wr[n,k-D]
// stored at ((k>>1)*D + n)*2 + (k&1)  (pairs of K contiguous per n)
__global__ __launch_bounds__(256) void pack_w_kernel(const float* __restrict__ wr,
                                                     const float* __restrict__ wi,
                                                     unsigned short* __restrict__ wpR,
                                                     unsigned short* __restrict__ wpI) {
    int idx = blockIdx.x * 256 + threadIdx.x;          // over K2*D
    int k = idx >> 10;
    int n = idx & (DIM - 1);
    float rv, iv;
    if (k < DIM) { rv = wr[n * DIM + k];          iv = wi[n * DIM + k]; }
    else         { rv = -wi[n * DIM + (k - DIM)]; iv = wr[n * DIM + (k - DIM)]; }
    size_t dst = ((size_t)(k >> 1) * DIM + n) * 2 + (k & 1);
    wpR[dst] = f2bf(rv);
    wpI[dst] = f2bf(iv);
}

__global__ __launch_bounds__(256) void pack_bias_kernel(const float* __restrict__ br,
                                                        const float* __restrict__ bi,
                                                        float* __restrict__ cbr,
                                                        float* __restrict__ cbi) {
    int n = blockIdx.x * 256 + threadIdx.x;
    if (n < DIM) { cbr[n] = br[n] - bi[n]; cbi[n] = br[n] + bi[n]; }
}

// ---------------------------------------------------------------------------
// WMMA GEMM: C[M,N] = A(bf16 row-major [M,K]) * Bp(pair-packed [K/2][N][2]) + bias
// block tile 64(M) x 128(N), 8 waves (2x4), wave tile 32x32, BK = 32
// ---------------------------------------------------------------------------
template <bool OUTF32>
__global__ __launch_bounds__(256) void gemm_bf16_wmma(const unsigned int* __restrict__ Ad,
                                                      const unsigned int* __restrict__ Bp,
                                                      const float* __restrict__ bias,
                                                      void* __restrict__ Cout,
                                                      int M, int N, int K) {
    __shared__ unsigned int a_lds[64 * 16];   // 64 rows x 16 dwords (32 bf16)
    __shared__ unsigned int b_lds[16 * 128];  // 16 k-pairs x 128 cols

    const int tid  = threadIdx.x;
    const int lane = tid & 31;
    const int wid  = tid >> 5;
    const int wm   = wid >> 2;        // 0..1
    const int wn   = wid & 3;         // 0..3
    const int m0   = blockIdx.y * 64;
    const int n0   = blockIdx.x * 128;
    const int hi   = (lane >= 16);

    v8f acc[2][2] = {};

    for (int k0 = 0; k0 < K; k0 += 32) {
#if USE_ASYNC_LDS
        {
            // A tile: 256 x b128 (one per thread)
            int row = tid >> 2, c4 = tid & 3;
            async_copy_b128(&Ad[(((size_t)(m0 + row) * K + k0) >> 1) + c4 * 4],
                            &a_lds[row * 16 + c4 * 4]);
            // B tile: 512 x b128 (two per thread)
#pragma unroll
            for (int i = 0; i < 2; ++i) {
                int t  = tid + i * 256;
                int kk = t >> 5;
                int n4 = t & 31;
                async_copy_b128(&Bp[(size_t)((k0 >> 1) + kk) * N + n0 + n4 * 4],
                                &b_lds[kk * 128 + n4 * 4]);
            }
        }
#else
        // ---- stage A tile: 64x32 bf16 = 1024 dwords ----
#pragma unroll
        for (int i = 0; i < 4; ++i) {
            int t   = tid + i * 256;
            int row = t >> 4;
            int c2  = t & 15;
            a_lds[t] = Ad[(((size_t)(m0 + row) * K + k0) >> 1) + c2];
        }
        // ---- stage B tile: 16x128 dwords ----
#pragma unroll
        for (int i = 0; i < 8; ++i) {
            int t  = tid + i * 256;
            int kk = t >> 7;
            int n  = t & 127;
            b_lds[t] = Bp[(size_t)((k0 >> 1) + kk) * N + n0 + n];
        }
#endif
        if (k0 + 32 < K) {
            // speculative prefetch of next B tile -> global_prefetch_b8
            __builtin_prefetch(&Bp[(size_t)((k0 >> 1) + 16) * N + n0 + (tid & 127)], 0, 1);
        }
#if USE_ASYNC_LDS
        async_wait0();
#endif
        __syncthreads();

        // ---- fragments ----
        v16bf afr[2], bfr[2];
#pragma unroll
        for (int mt = 0; mt < 2; ++mt) {
            U8 u;
            int mrow = wm * 32 + mt * 16 + (lane & 15);
#pragma unroll
            for (int v = 0; v < 8; ++v) {
                int kh = (hi ? 8 : 0) + v;  // (k>>1)
                u.x[v] = a_lds[mrow * 16 + kh];
            }
            afr[mt] = as_bf16x16(u);
        }
#pragma unroll
        for (int nt = 0; nt < 2; ++nt) {
            U8 u;
            int ncol = wn * 32 + nt * 16 + (lane & 15);
#pragma unroll
            for (int v = 0; v < 8; ++v) {
                u.x[v] = b_lds[(v + (hi ? 8 : 0)) * 128 + ncol];
            }
            bfr[nt] = as_bf16x16(u);
        }
#pragma unroll
        for (int mt = 0; mt < 2; ++mt)
#pragma unroll
            for (int nt = 0; nt < 2; ++nt)
                acc[mt][nt] = wmma_bf16(afr[mt], bfr[nt], acc[mt][nt]);
        __syncthreads();
    }

    // ---- epilogue: C layout: VGPR v -> row (v + 8*hi), lane&15 -> col ----
#pragma unroll
    for (int mt = 0; mt < 2; ++mt) {
#pragma unroll
        for (int nt = 0; nt < 2; ++nt) {
            int col = n0 + wn * 32 + nt * 16 + (lane & 15);
            float bv = bias[col];
#pragma unroll
            for (int v = 0; v < 8; ++v) {
                int row = m0 + wm * 32 + mt * 16 + v + (hi ? 8 : 0);
                float val = acc[mt][nt][v] + bv;
                if (OUTF32)
                    ((float*)Cout)[(size_t)row * N + col] = val;
                else
                    ((unsigned short*)Cout)[(size_t)row * N + col] = f2bf(val);
            }
        }
    }
}

// ---------------------------------------------------------------------------
// Flash attention over complex q/k (real part of q . conj(k)) with shared probs
// grid: B*H*(SEQ/64) blocks, 128 threads (4 waves x 16 query rows)
// ---------------------------------------------------------------------------
__global__ __launch_bounds__(128) void attn_kernel(const unsigned int* __restrict__ qr,
                                                   const unsigned int* __restrict__ qi,
                                                   const unsigned int* __restrict__ kr,
                                                   const unsigned int* __restrict__ ki,
                                                   const unsigned int* __restrict__ vr,
                                                   const unsigned int* __restrict__ vi,
                                                   unsigned short* __restrict__ ac) {
    __shared__ unsigned int   kr_lds[32 * 32];      // 32 keys x 32 dwords (64 bf16)
    __shared__ unsigned int   ki_lds[32 * 32];
    __shared__ unsigned short vr_lds[32 * 64];      // pair-interleaved by key
    __shared__ unsigned short vi_lds[32 * 64];
    __shared__ unsigned int   p_lds[4 * 16 * 16];   // per-wave 16x32 bf16

    const int tid  = threadIdx.x;
    const int lane = tid & 31;
    const int wid  = tid >> 5;
    const int hi   = (lane >= 16);
    const int h8   = hi ? 8 : 0;

    const int blk  = blockIdx.x;
    const int qblk = blk & 31;           // SEQ/64 = 32
    const int bh   = blk >> 5;
    const int b    = bh >> 4;
    const int h    = bh & 15;
    const int q0   = qblk * 64 + wid * 16;
    const int qrow = b * SEQ + q0 + (lane & 15);

    // Q fragments: [comp][dchunk], A-layout, dwords straight from global bf16
    v16bf aq[2][2];
#pragma unroll
    for (int comp = 0; comp < 2; ++comp) {
        const unsigned int* src = comp ? qi : qr;
#pragma unroll
        for (int dc = 0; dc < 2; ++dc) {
            U8 u;
#pragma unroll
            for (int v = 0; v < 8; ++v) {
                int kd = dc * 32 + (hi ? 16 : 0) + 2 * v;
                u.x[v] = src[(size_t)qrow * (DIM / 2) + h * (HDIM / 2) + (kd >> 1)];
            }
            aq[comp][dc] = as_bf16x16(u);
        }
    }

    float mrow[8], lrow[8];
#pragma unroll
    for (int v = 0; v < 8; ++v) { mrow[v] = -1e30f; lrow[v] = 0.0f; }
    v8f oacc[2][4] = {};

    const float scale = 0.125f;   // 1/sqrt(64)

    for (int kb = 0; kb < SEQ; kb += 32) {
        // ---- stage K blocks (row-major) ----
#if USE_ASYNC_LDS
#pragma unroll
        for (int i = 0; i < 2; ++i) {
            int u   = tid + i * 128;      // 256 b128 per matrix
            int key = u >> 3;
            int c4  = u & 7;
            size_t g = (size_t)(b * SEQ + kb + key) * (DIM / 2) + h * (HDIM / 2) + c4 * 4;
            async_copy_b128(&kr[g], &kr_lds[key * 32 + c4 * 4]);
            async_copy_b128(&ki[g], &ki_lds[key * 32 + c4 * 4]);
        }
#endif
        // ---- stage V blocks (pair-interleaved by key; needs manual repack) ----
#pragma unroll
        for (int i = 0; i < 8; ++i) {
            int t   = tid + i * 128;
            int key = t >> 5;
            int c   = t & 31;
            size_t g = (size_t)(b * SEQ + kb + key) * (DIM / 2) + h * (HDIM / 2) + c;
#if !USE_ASYNC_LDS
            kr_lds[t] = kr[g];
            ki_lds[t] = ki[g];
#endif
            int base = (key >> 1) * 128 + (key & 1);
            int d = c * 2;
            unsigned int wv = vr[g];
            vr_lds[base + d * 2]       = (unsigned short)(wv & 0xFFFFu);
            vr_lds[base + (d + 1) * 2] = (unsigned short)(wv >> 16);
            wv = vi[g];
            vi_lds[base + d * 2]       = (unsigned short)(wv & 0xFFFFu);
            vi_lds[base + (d + 1) * 2] = (unsigned short)(wv >> 16);
        }
#if USE_ASYNC_LDS
        async_wait0();
#endif
        __syncthreads();

        // ---- S tiles: S[q, key] = qr.kr + qi.ki, two 16-key tiles ----
        v8f s[2] = {};
#pragma unroll
        for (int kt = 0; kt < 2; ++kt) {
            int keyrow = kt * 16 + (lane & 15);
#pragma unroll
            for (int dc = 0; dc < 2; ++dc) {
                U8 ur, uii;
#pragma unroll
                for (int v = 0; v < 8; ++v) {
                    int dh = dc * 16 + (hi ? 8 : 0) + v;   // (d>>1)
                    ur.x[v]  = kr_lds[keyrow * 32 + dh];
                    uii.x[v] = ki_lds[keyrow * 32 + dh];
                }
                s[kt] = wmma_bf16(aq[0][dc], as_bf16x16(ur),  s[kt]);
                s[kt] = wmma_bf16(aq[1][dc], as_bf16x16(uii), s[kt]);
            }
        }
        s[0] *= scale;
        s[1] *= scale;

        // ---- online softmax: row v lives in lanes {0..15}+16*hi ----
        unsigned short* pw = (unsigned short*)&p_lds[wid * 256];
#pragma unroll
        for (int v = 0; v < 8; ++v) {
            float mx = fmaxf(s[0][v], s[1][v]);
#pragma unroll
            for (int off = 1; off < 16; off <<= 1)
                mx = fmaxf(mx, __shfl_xor(mx, off, 16));
            float mnew = fmaxf(mrow[v], mx);
            float p0 = __expf(s[0][v] - mnew);
            float p1 = __expf(s[1][v] - mnew);
            float ps = p0 + p1;
#pragma unroll
            for (int off = 1; off < 16; off <<= 1)
                ps += __shfl_xor(ps, off, 16);
            float sc = __expf(mrow[v] - mnew);
            lrow[v] = lrow[v] * sc + ps;
            mrow[v] = mnew;
#pragma unroll
            for (int comp = 0; comp < 2; ++comp)
#pragma unroll
                for (int nt = 0; nt < 4; ++nt)
                    oacc[comp][nt][v] *= sc;
            int m = v + h8;
            pw[m * 32 + (lane & 15)]      = f2bf(p0);
            pw[m * 32 + 16 + (lane & 15)] = f2bf(p1);
        }
        __syncthreads();

        // ---- P as A-fragment, PV accumulation ----
        const unsigned int* pd = &p_lds[wid * 256];
        U8 up;
#pragma unroll
        for (int v = 0; v < 8; ++v) {
            int kh = (hi ? 8 : 0) + v;  // (key>>1)
            up.x[v] = pd[(lane & 15) * 16 + kh];
        }
        v16bf pf = as_bf16x16(up);
        const unsigned int* vrd = (const unsigned int*)vr_lds;
        const unsigned int* vid = (const unsigned int*)vi_lds;
#pragma unroll
        for (int nt = 0; nt < 4; ++nt) {
            U8 ur, uii;
#pragma unroll
            for (int v = 0; v < 8; ++v) {
                int idx = (v + h8) * 64 + nt * 16 + (lane & 15);
                ur.x[v]  = vrd[idx];
                uii.x[v] = vid[idx];
            }
            oacc[0][nt] = wmma_bf16(pf, as_bf16x16(ur),  oacc[0][nt]);
            oacc[1][nt] = wmma_bf16(pf, as_bf16x16(uii), oacc[1][nt]);
        }
        __syncthreads();
    }

    // ---- normalize + write combined A = [ar | ai] (bf16, [ROWS x 2D]) ----
#pragma unroll
    for (int v = 0; v < 8; ++v) {
        float inv = 1.0f / lrow[v];
        int arow = b * SEQ + q0 + v + h8;
#pragma unroll
        for (int comp = 0; comp < 2; ++comp) {
#pragma unroll
            for (int nt = 0; nt < 4; ++nt) {
                int col = comp * DIM + h * HDIM + nt * 16 + (lane & 15);
                ac[(size_t)arow * K2 + col] = f2bf(oacc[comp][nt][v] * inv);
            }
        }
    }
}

// ---------------------------------------------------------------------------
// Launch
// ---------------------------------------------------------------------------
extern "C" void kernel_launch(void* const* d_in, const int* in_sizes, int n_in,
                              void* d_out, int out_size, void* d_ws, size_t ws_size,
                              hipStream_t stream) {
    (void)in_sizes; (void)n_in; (void)out_size; (void)ws_size;

    const float* z_real = (const float*)d_in[0];
    const float* z_imag = (const float*)d_in[1];
    // per projection: wr, wi, br, bi at d_in[2+4*p ...]

    char* ws = (char*)d_ws;
    size_t off = 0;
    unsigned short* zc = (unsigned short*)(ws + off); off += (size_t)ROWS * K2 * 2;
    unsigned short* wp[8];   // qR,qI,kR,kI,vR,vI,oR,oI
    for (int i = 0; i < 8; ++i) { wp[i] = (unsigned short*)(ws + off); off += (size_t)K2 * DIM * 2; }
    float* cb[8];            // combined biases, same order
    for (int i = 0; i < 8; ++i) { cb[i] = (float*)(ws + off); off += (size_t)DIM * 4; }
    unsigned short* proj[6]; // qr, qi, kr, ki, vr, vi
    for (int i = 0; i < 6; ++i) { proj[i] = (unsigned short*)(ws + off); off += (size_t)ROWS * DIM * 2; }
    unsigned short* acm = (unsigned short*)(ws + off); off += (size_t)ROWS * K2 * 2;

    // ---- packing ----
    pack_z_kernel<<<(ROWS * K2) / 256, 256, 0, stream>>>(z_real, z_imag, zc);
    for (int p = 0; p < 4; ++p) {
        const float* wr = (const float*)d_in[2 + 4 * p + 0];
        const float* wi = (const float*)d_in[2 + 4 * p + 1];
        const float* br = (const float*)d_in[2 + 4 * p + 2];
        const float* bi = (const float*)d_in[2 + 4 * p + 3];
        pack_w_kernel<<<(K2 * DIM) / 256, 256, 0, stream>>>(wr, wi, wp[2 * p], wp[2 * p + 1]);
        pack_bias_kernel<<<(DIM + 255) / 256, 256, 0, stream>>>(br, bi, cb[2 * p], cb[2 * p + 1]);
    }

    // ---- q/k/v complex projections: 6 GEMMs (bf16 out) ----
    dim3 gg(DIM / 128, ROWS / 64);
    for (int p = 0; p < 3; ++p) {
        gemm_bf16_wmma<false><<<gg, 256, 0, stream>>>(
            (const unsigned int*)zc, (const unsigned int*)wp[2 * p], cb[2 * p],
            proj[2 * p], ROWS, DIM, K2);
        gemm_bf16_wmma<false><<<gg, 256, 0, stream>>>(
            (const unsigned int*)zc, (const unsigned int*)wp[2 * p + 1], cb[2 * p + 1],
            proj[2 * p + 1], ROWS, DIM, K2);
    }

    // ---- flash attention ----
    attn_kernel<<<BATCH * NHEAD * (SEQ / 64), 128, 0, stream>>>(
        (const unsigned int*)proj[0], (const unsigned int*)proj[1],
        (const unsigned int*)proj[2], (const unsigned int*)proj[3],
        (const unsigned int*)proj[4], (const unsigned int*)proj[5], acm);

    // ---- output complex projection: 2 GEMMs (f32 out -> d_out [2,B,L,D]) ----
    float* out_r = (float*)d_out;
    float* out_i = out_r + (size_t)ROWS * DIM;
    gemm_bf16_wmma<true><<<gg, 256, 0, stream>>>(
        (const unsigned int*)acm, (const unsigned int*)wp[6], cb[6], out_r, ROWS, DIM, K2);
    gemm_bf16_wmma<true><<<gg, 256, 0, stream>>>(
        (const unsigned int*)acm, (const unsigned int*)wp[7], cb[7], out_i, ROWS, DIM, K2);
}
